// AttentionSequencePoolingLayer_55490977464677
// MI455X (gfx1250) — matrix operand order repaired and checked by reading, not compiled, not run on hardware
//
#include <hip/hip_runtime.h>
#include <hip/hip_bf16.h>
#include <math.h>

typedef __attribute__((ext_vector_type(16))) __bf16          v16bf;
typedef __attribute__((ext_vector_type(8)))  unsigned int    v8ui;
typedef __attribute__((ext_vector_type(8)))  float           v8f;
typedef __attribute__((ext_vector_type(2)))  __bf16          v2bf;

#define B_   4096
#define T_   200
#define D_   64
#define H1_  80
#define H2_  40
#define NTILES  13      // ceil(200/16)
#define NWAVES  4
#define NTHREADS 128

// pack two fp32 -> one u32 holding {hi:bf16(b), lo:bf16(a)}
__device__ __forceinline__ unsigned int pk_bf16(float a, float b) {
#if __has_builtin(__builtin_amdgcn_cvt_pk_bf16_f32)
    v2bf r = __builtin_amdgcn_cvt_pk_bf16_f32(a, b);
    union { v2bf v; unsigned int u; } x; x.v = r; return x.u;
#else
    // v_perm_b32: take high halves (truncate-round)
    return __builtin_amdgcn_perm(__float_as_uint(b), __float_as_uint(a), 0x07060302u);
#endif
}
__device__ __forceinline__ unsigned short f2bf(float f) {
    unsigned int u = __float_as_uint(f);
    u += 0x7FFFu + ((u >> 16) & 1u);           // round-to-nearest-even
    return (unsigned short)(u >> 16);
}
__device__ __forceinline__ float bf2f(unsigned short h) {
    return __uint_as_float(((unsigned int)h) << 16);
}
__device__ __forceinline__ v16bf as_bf16(v8ui u) {
    union { v8ui u; v16bf b; } x; x.u = u; return x.b;
}
__device__ __forceinline__ float4 mul4(float4 a, float4 b) {
    return make_float4(a.x * b.x, a.y * b.y, a.z * b.z, a.w * b.w);
}

__global__ __launch_bounds__(NTHREADS) void attn_pool_kernel(
    const float* __restrict__ q, const float* __restrict__ k,
    const unsigned char* __restrict__ km,
    const float* __restrict__ W1, const float* __restrict__ b1,
    const float* __restrict__ W2, const float* __restrict__ b2,
    const float* __restrict__ W3, const float* __restrict__ b3,
    float* __restrict__ out)
{
    __shared__ __align__(16) float          qs[D_];
    __shared__ __align__(16) unsigned short w1sw[4 * 5 * 32 * 16]; // B-frag swizzled W1' bf16
    __shared__ __align__(16) unsigned short w2sw[3 * 3 * 32 * 16]; // B-frag swizzled W2  bf16
    __shared__               float          qb[H1_];               // per-b folded layer1 bias
    __shared__               float          b2p[48];
    __shared__               float          w3s[H2_];
    __shared__               float          scores[T_];
    __shared__               float          red[NTHREADS];
    __shared__ __align__(16) float          kst[NWAVES * 16 * D_]; // staged k tiles (fp32)
    __shared__ __align__(16) unsigned short scr[NWAVES * 16 * 96]; // per-wave h1/h2 scratch

    const int b    = blockIdx.x;
    const int tid  = threadIdx.x;
    const int lane = tid & 31;
    const int wave = tid >> 5;

    // ---- per-block weight prep, directly into B-fragment layout ----
    if (tid < D_) qs[tid] = q[(size_t)b * D_ + tid];

    // W1' folded: rows 0..63 = W1[64..127]-W1[128..191]; rows 64..127 = W1[192..255]
    for (int i = tid; i < 4 * 5 * 32 * 16; i += NTHREADS) {
        int e  = i & 15;
        int ln = (i >> 4) & 31;
        int f  = i >> 9;              // kf*5 + nt
        int kf = f / 5, nt = f % 5;
        int kk = kf * 32 + ((ln & 16) ? 8 : 0) + (e < 8 ? e : e + 8);
        int n  = nt * 16 + (ln & 15);
        float v = (kk < 64) ? W1[(64 + kk) * H1_ + n] - W1[(128 + kk) * H1_ + n]
                            : W1[(192 + (kk - 64)) * H1_ + n];
        w1sw[i] = f2bf(v);
    }
    // W2 padded to [96 x 48]
    for (int i = tid; i < 3 * 3 * 32 * 16; i += NTHREADS) {
        int e  = i & 15;
        int ln = (i >> 4) & 31;
        int f  = i >> 9;              // kf*3 + nt
        int kf = f / 3, nt = f % 3;
        int kk = kf * 32 + ((ln & 16) ? 8 : 0) + (e < 8 ? e : e + 8);
        int n  = nt * 16 + (ln & 15);
        float v = (kk < H1_ && n < H2_) ? W2[kk * H2_ + n] : 0.f;
        w2sw[i] = f2bf(v);
    }
    if (tid < 48)  b2p[tid] = (tid < H2_) ? b2[tid] : 0.f;
    if (tid < H2_) w3s[tid] = W3[tid];
    if (tid < H1_) {
        float s = b1[tid];
        for (int d = 0; d < D_; ++d)
            s += q[(size_t)b * D_ + d] * (W1[d * H1_ + tid] + W1[(128 + d) * H1_ + tid]);
        qb[tid] = s;
    }
    __syncthreads();

    const float b3v = b3[0];
    float*          kw  = &kst[wave * 16 * D_];
    unsigned short* h1s = &scr[wave * 16 * 96];
    unsigned short* h2s = h1s;                   // reused after A2 frags consumed
    const v8ui*     w1f = (const v8ui*)w1sw;
    const v8ui*     w2f = (const v8ui*)w2sw;

    const int m   = lane & 15;
    const int lh8 = (lane & 16) ? 8 : 0;

    for (int tile = wave; tile < NTILES; tile += NWAVES) {
        const int t0 = tile * 16;

        // stage this wave's 16x64 k tile into LDS (zero rows past T)
        for (int j = lane; j < 16 * 16; j += 32) {
            int row = j >> 4, c4 = j & 15;
            int t = t0 + row;
            float4 v = make_float4(0.f, 0.f, 0.f, 0.f);
            if (t < T_) v = *(const float4*)&k[(size_t)b * T_ * D_ + (size_t)t * D_ + c4 * 4];
            *(float4*)&kw[row * D_ + c4 * 4] = v;
        }
        // (re)zero h1 pad columns 80..95 (clobbered by h2 reuse last tile)
        for (int j = lane; j < 16 * 16; j += 32) {
            int row = j >> 4, c = 80 + (j & 15);
            h1s[row * 96 + c] = 0;
        }

        // ---- layer-1 A fragments: [16 x 128] = [k | q*k], packed bf16 ----
        v8ui a1[4];
        {
            const float4* krow4 = (const float4*)&kw[m * D_];
            const float4* q4    = (const float4*)qs;
            #pragma unroll
            for (int kf = 0; kf < 4; ++kf) {
                const int fbase = (kf < 2) ? kf * 32 : (kf - 2) * 32;   // float index base
                const int o = (fbase + lh8) >> 2;                       // float4 index
                float4 k0 = krow4[o],     k1 = krow4[o + 1];            // e = 0..7
                float4 k2 = krow4[o + 4], k3 = krow4[o + 5];            // e = 8..15
                if (kf >= 2) {
                    k0 = mul4(k0, q4[o]);     k1 = mul4(k1, q4[o + 1]);
                    k2 = mul4(k2, q4[o + 4]); k3 = mul4(k3, q4[o + 5]);
                }
                v8ui a;
                a[0] = pk_bf16(k0.x, k0.y); a[1] = pk_bf16(k0.z, k0.w);
                a[2] = pk_bf16(k1.x, k1.y); a[3] = pk_bf16(k1.z, k1.w);
                a[4] = pk_bf16(k2.x, k2.y); a[5] = pk_bf16(k2.z, k2.w);
                a[6] = pk_bf16(k3.x, k3.y); a[7] = pk_bf16(k3.z, k3.w);
                a1[kf] = a;
            }
        }

        // ---- layer 1: [16x128] @ [128x80] + qbias, relu ----
        #pragma unroll
        for (int nt = 0; nt < 5; ++nt) {
            const int n = nt * 16 + m;
            float cinit = qb[n];
            v8f acc;
            #pragma unroll
            for (int r = 0; r < 8; ++r) acc[r] = cinit;
            #pragma unroll
            for (int kf = 0; kf < 4; ++kf) {
                v8ui bf = w1f[(kf * 5 + nt) * 32 + lane];      // contiguous 32B frag
                acc = __builtin_amdgcn_wmma_f32_16x16x32_bf16(
                        false, as_bf16(a1[kf]), false, as_bf16(bf),
                        (short)0, acc, false, false);
            }
            #pragma unroll
            for (int r = 0; r < 8; r += 2) {
                unsigned int p = pk_bf16(fmaxf(acc[r], 0.f), fmaxf(acc[r + 1], 0.f));
                h1s[(r + lh8) * 96 + n]     = (unsigned short)p;
                h1s[(r + 1 + lh8) * 96 + n] = (unsigned short)(p >> 16);
            }
        }

        // ---- layer-2 A fragments from h1s (two aligned 16B runs each) ----
        v8ui a2[3];
        #pragma unroll
        for (int kf = 0; kf < 3; ++kf) {
            const int base = kf * 32 + lh8;
            union { uint4 u[2]; v8ui v; } x;
            x.u[0] = *(const uint4*)&h1s[m * 96 + base];        // e = 0..7
            x.u[1] = *(const uint4*)&h1s[m * 96 + base + 16];   // e = 8..15
            a2[kf] = x.v;
        }

        // ---- layer 2: [16x96] @ [96x48] + b2, relu ----
        #pragma unroll
        for (int nt = 0; nt < 3; ++nt) {
            const int n = nt * 16 + m;
            float cinit = b2p[n];
            v8f acc;
            #pragma unroll
            for (int r = 0; r < 8; ++r) acc[r] = cinit;
            #pragma unroll
            for (int kf = 0; kf < 3; ++kf) {
                v8ui bf = w2f[(kf * 3 + nt) * 32 + lane];
                acc = __builtin_amdgcn_wmma_f32_16x16x32_bf16(
                        false, as_bf16(a2[kf]), false, as_bf16(bf),
                        (short)0, acc, false, false);
            }
            #pragma unroll
            for (int r = 0; r < 8; r += 2) {
                unsigned int p = pk_bf16(fmaxf(acc[r], 0.f), fmaxf(acc[r + 1], 0.f));
                h2s[(r + lh8) * 48 + n]     = (unsigned short)p;
                h2s[(r + 1 + lh8) * 48 + n] = (unsigned short)(p >> 16);
            }
        }

        // ---- layer 3 (40 -> 1) + mask ----
        if (lane < 16) {
            int t = t0 + lane;
            if (t < T_) {
                float s = b3v;
                const unsigned int* row32 = (const unsigned int*)&h2s[lane * 48];
                #pragma unroll
                for (int n2 = 0; n2 < H2_ / 2; ++n2) {
                    unsigned int p = row32[n2];
                    s += bf2f((unsigned short)p)         * w3s[2 * n2]
                       + bf2f((unsigned short)(p >> 16)) * w3s[2 * n2 + 1];
                }
                bool mk = km[(size_t)b * T_ + t] != 0;
                scores[t] = mk ? s : -INFINITY;
            }
        }
    }
    __syncthreads();

    // ---- masked softmax over T ----
    float lm = -INFINITY;
    for (int t = tid; t < T_; t += NTHREADS) lm = fmaxf(lm, scores[t]);
    red[tid] = lm;
    __syncthreads();
    for (int s = NTHREADS / 2; s > 0; s >>= 1) {
        if (tid < s) red[tid] = fmaxf(red[tid], red[tid + s]);
        __syncthreads();
    }
    const float mx = red[0];
    __syncthreads();

    float ls = 0.f;
    for (int t = tid; t < T_; t += NTHREADS) {
        float e = __expf(scores[t] - mx);
        scores[t] = e;
        ls += e;
    }
    red[tid] = ls;
    __syncthreads();
    for (int s = NTHREADS / 2; s > 0; s >>= 1) {
        if (tid < s) red[tid] += red[tid + s];
        __syncthreads();
    }
    const float inv = 1.f / red[0];
    __syncthreads();

    // ---- weighted sum: o[b,d] = sum_t w[t] * k[b,t,d] (coalesced, L2-hit) ----
    {
        const int d = tid & (D_ - 1);
        const int g = tid >> 6;                 // 0..1, 100 t's each
        float acc = 0.f;
        for (int t = g * 100; t < g * 100 + 100; ++t)
            acc += scores[t] * k[(size_t)b * T_ * D_ + (size_t)t * D_ + d];
        red[tid] = acc;
        __syncthreads();
        if (tid < D_)
            out[(size_t)b * D_ + tid] = (red[tid] + red[tid + D_]) * inv;
    }
}

extern "C" void kernel_launch(void* const* d_in, const int* in_sizes, int n_in,
                              void* d_out, int out_size, void* d_ws, size_t ws_size,
                              hipStream_t stream) {
    const float*         q  = (const float*)d_in[0];
    const float*         k  = (const float*)d_in[1];
    const unsigned char* km = (const unsigned char*)d_in[2];   // jnp.bool_ = 1 byte
    const float*         W1 = (const float*)d_in[3];
    const float*         b1 = (const float*)d_in[4];
    const float*         W2 = (const float*)d_in[5];
    const float*         b2 = (const float*)d_in[6];
    const float*         W3 = (const float*)d_in[7];
    const float*         b3 = (const float*)d_in[8];
    float* out = (float*)d_out;

    attn_pool_kernel<<<B_, NTHREADS, 0, stream>>>(q, k, km, W1, b1, W2, b2, W3, b3, out);
}
